// Qwen3MoeMLPStack_74457553043825
// MI455X (gfx1250) — matrix operand order repaired and checked by reading, not compiled, not run on hardware
//
#include <hip/hip_runtime.h>
#include <hip/hip_bf16.h>

// ---------------------------------------------------------------------------
// Grouped-GEMM SwiGLU MoE MLP for MI455X (gfx1250, wave32, WMMA bf16).
//   h   = silu(x @ gate[e]) * (x @ up[e])      (kernel 1, h stored bf16 in ws)
//   out = h @ down[e]                          (kernel 2)
// f32 -> bf16 conversion while staging into LDS; f32 accumulation via
// v_wmma_f32_16x16x32_bf16. Register-staged pipeline overlaps the next
// K-tile's global_load_b128 with the current tile's WMMA work. All per-lane
// global offsets are 32-bit so loads use SGPR-base + VGPR-offset form.
// Per-expert working set (~20 MB) lives in the 192 MB L2, so HBM traffic is
// near-compulsory (~1.1 GB -> ~50 us at 23.3 TB/s, balanced vs bf16 WMMA).
// ---------------------------------------------------------------------------

typedef __attribute__((ext_vector_type(16))) __bf16 v16bf;
typedef __attribute__((ext_vector_type(8)))  __bf16 v8bf;
typedef __attribute__((ext_vector_type(4)))  __bf16 v4bf;
typedef __attribute__((ext_vector_type(8)))  float  v8f;
typedef __attribute__((ext_vector_type(4)))  float  v4f;

#define NE 32
#define NT 32768
#define NH 2048
#define NI 768

#define BM 128          // block tile rows (tokens)
#define BN1 64          // kernel-1 block cols (gate+up fused -> reg pressure)
#define BN2 128         // kernel-2 block cols (single B -> bigger tile)
#define BK 32           // K step == WMMA K
#define LDS_STRIDE 40   // padded element stride (80 B, keeps 16B alignment)

__device__ __forceinline__ int expert_of_row(const int* __restrict__ gs, int row) {
    int off = 0;
    #pragma unroll
    for (int e = 0; e < NE; ++e) {
        int nxt = off + gs[e];
        if (row < nxt) return e;
        off = nxt;
    }
    return NE - 1;
}

__device__ __forceinline__ v16bf concat8(v8bf lo, v8bf hi) {
    return __builtin_shufflevector(lo, hi, 0,1,2,3,4,5,6,7,8,9,10,11,12,13,14,15);
}

// A fragment (16x32 bf16, M x K): lane l -> row (l&15); lanes 0-15 hold
// K {0..7,16..23}, lanes 16-31 hold K {8..15,24..31}; 2 values per VGPR.
__device__ __forceinline__ v16bf load_a_frag(const __bf16* __restrict__ lds,
                                             int lane, int row_base) {
    int r  = row_base + (lane & 15);
    int kg = lane >> 4;
    const v8bf* p0 = (const v8bf*)(lds + r * LDS_STRIDE + kg * 8);
    const v8bf* p1 = (const v8bf*)(lds + r * LDS_STRIDE + 16 + kg * 8);
    return concat8(*p0, *p1);
}

// B fragment (32x16 bf16, K x N): LDS holds B transposed [N][K]; lane l ->
// col (l&15); lanes 0-15 hold K {0..7,16..23}, lanes 16-31 hold K {8..15,24..31}.
__device__ __forceinline__ v16bf load_b_frag(const __bf16* __restrict__ lds,
                                             int lane, int col_base) {
    int c  = col_base + (lane & 15);
    int kh = lane >> 4;
    const v8bf* p0 = (const v8bf*)(lds + c * LDS_STRIDE + kh * 8);
    const v8bf* p1 = (const v8bf*)(lds + c * LDS_STRIDE + 16 + kh * 8);
    return concat8(*p0, *p1);
}

__device__ __forceinline__ v8f wmma_bf16(v16bf a, v16bf b, v8f c) {
    return __builtin_amdgcn_wmma_f32_16x16x32_bf16(
        false, a, false, b, (short)0, c, false, false);
}

__device__ __forceinline__ v4bf cvt4(v4f v) {
    v4bf t;
    #pragma unroll
    for (int j = 0; j < 4; ++j) t[j] = (__bf16)v[j];
    return t;
}

// ---------------------------------------------------------------------------
// Kernel 1: fused gate+up GEMM + SwiGLU -> h (bf16) in workspace.
// grid = (NI/BN1, NT/BM), block = 256 (8 waves; 4x2 grid, 32x32 per wave)
// ---------------------------------------------------------------------------
__global__ __launch_bounds__(256)
void moe_gateup_kernel(const float* __restrict__ x,     // [T,H]
                       const float* __restrict__ gk,    // [E,H,I]
                       const float* __restrict__ uk,    // [E,H,I]
                       const int*   __restrict__ gs,    // [E]
                       __bf16*      __restrict__ hws)   // [T,I] bf16
{
    __shared__ __bf16 sA [BM  * LDS_STRIDE];
    __shared__ __bf16 sBg[BN1 * LDS_STRIDE];
    __shared__ __bf16 sBu[BN1 * LDS_STRIDE];

    const int tid    = threadIdx.x;
    const int lane   = tid & 31;
    const int wid    = tid >> 5;
    const int wave_n = wid & 1;      // 0..1 (32 cols each)
    const int wave_m = wid >> 1;     // 0..3 (32 rows each)

    const int n0 = blockIdx.x * BN1;
    const int m0 = blockIdx.y * BM;
    const int e  = expert_of_row(gs, m0);

    const float* __restrict__ xb = x  + (size_t)m0 * NH;
    const float* __restrict__ gb = gk + (size_t)e * NH * NI + n0;
    const float* __restrict__ ub = uk + (size_t)e * NH * NI + n0;

    v8f accg[2][2] = {};
    v8f accu[2][2] = {};

    v4f aR[4], gR[2], uR[2];

    auto load_tile = [&](int k0) {
        #pragma unroll
        for (int i = 0; i < 4; ++i) {                 // A: 128x32, 4 float4/thr
            int idx4 = i * 256 + tid;
            int r = idx4 >> 3, c4 = idx4 & 7;
            aR[i] = *(const v4f*)(xb + r * NH + k0 + c4 * 4);
        }
        #pragma unroll
        for (int i = 0; i < 2; ++i) {                 // B: 32x64, 2 float4/thr
            int idx4 = i * 256 + tid;
            int k = idx4 >> 4, n4 = idx4 & 15;
            int off = (k0 + k) * NI + n4 * 4;
            gR[i] = *(const v4f*)(gb + off);
            uR[i] = *(const v4f*)(ub + off);
        }
    };
    auto store_tile = [&]() {
        #pragma unroll
        for (int i = 0; i < 4; ++i) {
            int idx4 = i * 256 + tid;
            int r = idx4 >> 3, c4 = idx4 & 7;
            *(v4bf*)(sA + r * LDS_STRIDE + c4 * 4) = cvt4(aR[i]);
        }
        #pragma unroll
        for (int i = 0; i < 2; ++i) {                 // LDS transpose for B
            int idx4 = i * 256 + tid;
            int k = idx4 >> 4, n4 = idx4 & 15;
            #pragma unroll
            for (int j = 0; j < 4; ++j) {
                sBg[(n4 * 4 + j) * LDS_STRIDE + k] = (__bf16)gR[i][j];
                sBu[(n4 * 4 + j) * LDS_STRIDE + k] = (__bf16)uR[i][j];
            }
        }
    };

    load_tile(0);
    for (int k0 = 0; k0 < NH; k0 += BK) {
        store_tile();
        __syncthreads();
        if (k0 + BK < NH) load_tile(k0 + BK);         // overlap with WMMA below

        v16bf afrag[2];
        #pragma unroll
        for (int ms = 0; ms < 2; ++ms)
            afrag[ms] = load_a_frag(sA, lane, wave_m * 32 + ms * 16);

        #pragma unroll
        for (int ns = 0; ns < 2; ++ns) {
            v16bf bg = load_b_frag(sBg, lane, wave_n * 32 + ns * 16);
            v16bf bu = load_b_frag(sBu, lane, wave_n * 32 + ns * 16);
            #pragma unroll
            for (int ms = 0; ms < 2; ++ms) {
                accg[ms][ns] = wmma_bf16(afrag[ms], bg, accg[ms][ns]);
                accu[ms][ns] = wmma_bf16(afrag[ms], bu, accu[ms][ns]);
            }
        }
        __syncthreads();
    }

    // ---- epilogue: h = silu(gate) * up, store bf16 ----
    #pragma unroll
    for (int ms = 0; ms < 2; ++ms) {
        #pragma unroll
        for (int ns = 0; ns < 2; ++ns) {
            int n  = n0 + wave_n * 32 + ns * 16 + (lane & 15);
            int mb = m0 + wave_m * 32 + ms * 16 + (lane >> 4) * 8;
            v8f g = accg[ms][ns];
            v8f u = accu[ms][ns];
            #pragma unroll
            for (int r = 0; r < 8; ++r) {
                float gv = g[r];
                float hv = (gv / (1.0f + __expf(-gv))) * u[r];
                hws[(size_t)(mb + r) * NI + n] = (__bf16)hv;
            }
        }
    }
}

// ---------------------------------------------------------------------------
// Kernel 2: out = h @ down[e].  grid = (NH/BN2, NT/BM), block = 256.
// 8 waves; wave grid 4(M) x 2(N); wave tile 32 x 64 (2x4 WMMA accumulators).
// Per K-step: 12 ds_load_b128 -> 8 WMMA.
// ---------------------------------------------------------------------------
__global__ __launch_bounds__(256)
void moe_down_kernel(const __bf16* __restrict__ hws,   // [T,I] bf16
                     const float*  __restrict__ dk,    // [E,I,H]
                     const int*    __restrict__ gs,    // [E]
                     float*        __restrict__ out)   // [T,H]
{
    __shared__ __bf16 sA[BM  * LDS_STRIDE];
    __shared__ __bf16 sB[BN2 * LDS_STRIDE];

    const int tid    = threadIdx.x;
    const int lane   = tid & 31;
    const int wid    = tid >> 5;
    const int wave_n = wid & 1;      // 0..1 (64 cols each)
    const int wave_m = wid >> 1;     // 0..3 (32 rows each)

    const int n0 = blockIdx.x * BN2;
    const int m0 = blockIdx.y * BM;
    const int e  = expert_of_row(gs, m0);

    const __bf16* __restrict__ hb = hws + (size_t)m0 * NI;
    const float*  __restrict__ db = dk + (size_t)e * NI * NH + n0;

    v8f acc[2][4] = {};

    v8bf aR[2];   // A already bf16: 2 x 16B loads/thread (global_load_b128)
    v4f  bR[4];   // B: 32x128 f32, 4 float4/thread

    auto load_tile = [&](int k0) {
        #pragma unroll
        for (int i = 0; i < 2; ++i) {                 // A: 128x32 bf16
            int idx8 = i * 256 + tid;
            int r = idx8 >> 2, c8 = idx8 & 3;
            aR[i] = *(const v8bf*)(hb + r * NI + k0 + c8 * 8);
        }
        #pragma unroll
        for (int i = 0; i < 4; ++i) {                 // B: 32x128 f32
            int idx4 = i * 256 + tid;
            int k = idx4 >> 5, n4 = idx4 & 31;        // 32 float4 per row
            bR[i] = *(const v4f*)(db + (k0 + k) * NH + n4 * 4);
        }
    };
    auto store_tile = [&]() {
        #pragma unroll
        for (int i = 0; i < 2; ++i) {
            int idx8 = i * 256 + tid;
            int r = idx8 >> 2, c8 = idx8 & 3;
            *(v8bf*)(sA + r * LDS_STRIDE + c8 * 8) = aR[i];        // ds_store_b128
        }
        #pragma unroll
        for (int i = 0; i < 4; ++i) {
            int idx4 = i * 256 + tid;
            int k = idx4 >> 5, n4 = idx4 & 31;
            #pragma unroll
            for (int j = 0; j < 4; ++j)
                sB[(n4 * 4 + j) * LDS_STRIDE + k] = (__bf16)bR[i][j];
        }
    };

    load_tile(0);
    for (int k0 = 0; k0 < NI; k0 += BK) {
        store_tile();
        __syncthreads();
        if (k0 + BK < NI) load_tile(k0 + BK);

        v16bf afrag[2];
        #pragma unroll
        for (int ms = 0; ms < 2; ++ms)
            afrag[ms] = load_a_frag(sA, lane, wave_m * 32 + ms * 16);

        #pragma unroll
        for (int ns = 0; ns < 4; ++ns) {
            v16bf b = load_b_frag(sB, lane, wave_n * 64 + ns * 16);
            #pragma unroll
            for (int ms = 0; ms < 2; ++ms)
                acc[ms][ns] = wmma_bf16(afrag[ms], b, acc[ms][ns]);
        }
        __syncthreads();
    }

    #pragma unroll
    for (int ms = 0; ms < 2; ++ms) {
        #pragma unroll
        for (int ns = 0; ns < 4; ++ns) {
            int n  = n0 + wave_n * 64 + ns * 16 + (lane & 15);
            int mb = m0 + wave_m * 32 + ms * 16 + (lane >> 4) * 8;
            v8f a = acc[ms][ns];
            #pragma unroll
            for (int r = 0; r < 8; ++r)
                out[(size_t)(mb + r) * NH + n] = a[r];
        }
    }
}

// ---------------------------------------------------------------------------
extern "C" void kernel_launch(void* const* d_in, const int* in_sizes, int n_in,
                              void* d_out, int out_size, void* d_ws, size_t ws_size,
                              hipStream_t stream) {
    const float* x  = (const float*)d_in[0];   // hidden_states [T,H]
    const float* gk = (const float*)d_in[1];   // gate_kernel   [E,H,I]
    const float* uk = (const float*)d_in[2];   // up_kernel     [E,H,I]
    const float* dk = (const float*)d_in[3];   // down_kernel   [E,I,H]
    const int*   gs = (const int*)d_in[4];     // group_sizes   [E]
    float* out = (float*)d_out;                // [T,H]

    __bf16* hws = (__bf16*)d_ws;               // h intermediate: T*I*2 = 48 MB

    dim3 block(256);
    dim3 grid1(NI / BN1, NT / BM);             // 12 x 256
    moe_gateup_kernel<<<grid1, block, 0, stream>>>(x, gk, uk, gs, hws);

    dim3 grid2(NH / BN2, NT / BM);             // 16 x 256
    moe_down_kernel<<<grid2, block, 0, stream>>>(hws, dk, gs, out);
}
